// Mapping_46892452938219
// MI455X (gfx1250) — matrix operand order repaired and checked by reading, not compiled, not run on hardware
//
#include <hip/hip_runtime.h>
#include <cstdint>
#include <cstddef>

#define DEVINL __device__ __forceinline__

typedef __attribute__((ext_vector_type(8)))  float        v8f;
typedef __attribute__((ext_vector_type(16))) __bf16       v16bf;
typedef __attribute__((ext_vector_type(4)))  unsigned int v4u;
typedef __attribute__((ext_vector_type(4)))  int          v4i;
typedef __attribute__((ext_vector_type(8)))  int          v8i;

union ABfrag { v16bf v; uint4 q[2]; };

// ---------------- geometry ----------------
static constexpr int BATCH = 4096;
// packed K per group: CIN * {9,5,9}
DEVINL int g_K(int g)    { return (g == 1) ? 320 : 576; }
DEVINL int g_npos(int g) { return (g == 1) ? 5 : 9; }
DEVINL int g_base(int g) { // halves offset of group g inside packed-W5 buffer
    return (g == 0) ? 0 : (g == 1) ? 9*576*256 : (9*576*256 + 9*320*256);
}
__device__ const int OFF5[3][9] = {
    {6,7,8,11,12,13,16,17,18},   // circle: 3x3 center block
    {0,4,20,24,12, 0,0,0,0},     // corner: 4 corners + center
    {1,3,5,9,15,19,21,23,12}};   // horse

// ---------------- workspace layout (bytes) ----------------
static constexpr size_t XB_OFF    = 0;
static constexpr size_t XB_BYTES  = (size_t)BATCH * 3136 * 2;               // bf16 x
static constexpr size_t W5B_OFF   = XB_OFF + XB_BYTES;
static constexpr size_t W5B_BYTES = (size_t)(9*576*256 + 9*320*256 + 9*576*256) * 2;
static constexpr size_t W1B_OFF   = W5B_OFF + W5B_BYTES;
static constexpr size_t WB_BYTES  = (size_t)27 * 256 * 256 * 2;
static constexpr size_t W2B_OFF   = W1B_OFF + WB_BYTES;
static constexpr size_t COPY_OFF  = W2B_OFF + WB_BYTES;
static constexpr size_t ACC_BYTES = (size_t)BATCH * 256 * 4;
static constexpr size_t JUMP_OFF  = COPY_OFF + ACC_BYTES;

// ---------------- helpers ----------------
DEVINL unsigned f2bf(float f) {
    unsigned u = __float_as_uint(f);
    unsigned r = u + 0x7FFFu + ((u >> 16) & 1u);     // round-to-nearest-even
    if ((u & 0x7F800000u) == 0x7F800000u) r = u;     // inf/nan passthrough
    return r >> 16;
}
DEVINL float mishf(float x) {
    float sp = (x > 20.f) ? x : log1pf(__expf(x));   // stable softplus
    return x * tanhf(sp);
}

// ---- TDM: async 1-D tile copy global -> LDS (D# per CDNA5 ISA) ----
DEVINL void tdm_issue(const unsigned short* src, unsigned lds_addr_bytes, unsigned nbytes) {
    unsigned long long ga = (unsigned long long)(uintptr_t)src;
    unsigned n8 = nbytes >> 3;                         // 8-byte elements
    v4u g0 = { 1u,                                     // count=1: one valid descriptor
               lds_addr_bytes,                         // lds_addr
               (unsigned)ga,                           // global_addr[31:0]
               (unsigned)((ga >> 32) & 0x01FFFFFFu) | (2u << 30) }; // addr[56:32] | type=2
    v8i g1 = { (int)(3u << 16),                        // data_size = 8B
               (int)((n8 & 0xFFFFu) << 16),            // tensor_dim0[15:0]
               (int)(n8 >> 16),                        // tensor_dim0[31:16]; tensor_dim1=0
               (int)((n8 & 0xFFFFu) << 16),            // tile_dim0 = n8 (1-D tile)
               0,                                      // tile_dim1 = tile_dim2 = 0
               (int)n8,                                // tensor_dim0_stride[31:0]
               0, 0 };
    v4i z4 = {0, 0, 0, 0};
#if defined(__clang_major__) && __clang_major__ >= 23
    v8i z8 = {0, 0, 0, 0, 0, 0, 0, 0};
    __builtin_amdgcn_tensor_load_to_lds(g0, g1, z4, z4, z8, 0);
#else
    __builtin_amdgcn_tensor_load_to_lds(g0, g1, z4, z4, 0);
#endif
}

DEVINL unsigned lds_addr_of(const void* p) { return (unsigned)(uintptr_t)p; }

DEVINL v8f wmma_bf16(const ABfrag& a, const ABfrag& b, v8f c) {
    return __builtin_amdgcn_wmma_f32_16x16x32_bf16(
        false, a.v, false, b.v, (short)0, c, false, false);
}

// 16 WMMAs over one 32-K chunk: 2 A-fragments (32 rows) x 8 B-fragments (128 cols).
// Each B fragment feeds two back-to-back WMMAs -> LDS latency amortized.
DEVINL void gemm2x8(v8f (*acc)[8], const unsigned short* a0, const unsigned short* a1,
                    const unsigned short* bb, int lane, int nbase) {
    ABfrag af0, af1;
    af0.q[0] = ((const uint4*)a0)[0];
    af0.q[1] = ((const uint4*)(a0 + 8))[0];
    af1.q[0] = ((const uint4*)a1)[0];
    af1.q[1] = ((const uint4*)(a1 + 8))[0];
    ABfrag bf[2];
    { const uint4* p = (const uint4*)&bb[(nbase * 32 + lane) * 16];
      bf[0].q[0] = p[0]; bf[0].q[1] = p[1]; }
    #pragma unroll
    for (int nt = 0; nt < 8; ++nt) {
        if (nt + 1 < 8) {   // prefetch next B fragment
            const uint4* p = (const uint4*)&bb[((nbase + nt + 1) * 32 + lane) * 16];
            bf[(nt + 1) & 1].q[0] = p[0]; bf[(nt + 1) & 1].q[1] = p[1];
        }
        acc[0][nt] = wmma_bf16(af0, bf[nt & 1], acc[0][nt]);
        acc[1][nt] = wmma_bf16(af1, bf[nt & 1], acc[1][nt]);
    }
}

// ---------------- prep kernels ----------------
__global__ void zero_acc_kernel(float* copyA, float* jumpA) {
    int i = blockIdx.x * 256 + threadIdx.x;
    copyA[i] = 0.f; jumpA[i] = 0.f;
}

__global__ void prep_x_kernel(const float4* __restrict__ x, uint2* __restrict__ xb) {
    size_t i = (size_t)blockIdx.x * 256 + threadIdx.x;  // 3,211,264 float4s
    float4 v = x[i];
    uint2 o;
    o.x = f2bf(v.x) | (f2bf(v.y) << 16);
    o.y = f2bf(v.z) | (f2bf(v.w) << 16);
    xb[i] = o;
}

// mask-packed W5 -> bf16, B-fragment-major: [gp][ktile][ntile][lane][16]
__global__ void prep_w5_kernel(const float* __restrict__ W5, unsigned short* __restrict__ w5b) {
    int gp = blockIdx.y, g = gp / 9, p = gp - g * 9;
    int Kg = g_K(g);
    int t  = blockIdx.x * 256 + threadIdx.x;
    if (t >= Kg * 256) return;
    int ktile = t >> 13, rem = t & 8191;
    int ntile = rem >> 9; rem &= 511;
    int lane  = rem >> 4; int h = rem & 15;
    int k = ktile * 32 + ((lane >> 4) << 4) + h;     // packed K index
    int n = ntile * 16 + (lane & 15);
    int c, pos;
    if (g == 1) { c = k / 5; pos = k - c * 5; } else { c = k / 9; pos = k - c * 9; }
    int korig = c * 25 + OFF5[g][pos];
    float v = W5[(size_t)gp * 409600 + (size_t)n * 1600 + korig];
    w5b[(size_t)g_base(g) + (size_t)p * Kg * 256 + t] = (unsigned short)f2bf(v);
}

// W1/W2 -> bf16, B-fragment-major; B(k,n) = W[n][k]
__global__ void prep_w12_kernel(const float* __restrict__ W1, const float* __restrict__ W2,
                                unsigned short* __restrict__ w1b, unsigned short* __restrict__ w2b) {
    int gp = blockIdx.y;
    int t  = blockIdx.x * 256 + threadIdx.x;         // < 65536
    int ktile = t >> 13, rem = t & 8191;
    int ntile = rem >> 9; rem &= 511;
    int lane  = rem >> 4; int h = rem & 15;
    int k = ktile * 32 + ((lane >> 4) << 4) + h;
    int n = ntile * 16 + (lane & 15);
    size_t src = (size_t)gp * 65536 + (size_t)n * 256 + k;
    size_t dst = (size_t)gp * 65536 + t;
    w1b[dst] = (unsigned short)f2bf(W1[src]);
    w2b[dst] = (unsigned short)f2bf(W2[src]);
}

// ---------------- fused main kernel ----------------
// grid: (32 row-tiles of 128, 27 (g,p)); block: 256 = 8 waves in a 4x2 grid;
// each wave owns a 32-row x 128-col sub-tile of the 128x256 block tile.
__global__ void __launch_bounds__(256)
fused_branch_kernel(const unsigned short* __restrict__ xb,
                    const unsigned short* __restrict__ w5b,
                    const unsigned short* __restrict__ w1b,
                    const unsigned short* __restrict__ w2b,
                    const float* __restrict__ b5, const float* __restrict__ gamma,
                    const float* __restrict__ beta, const float* __restrict__ rmean,
                    const float* __restrict__ rvar, const float* __restrict__ b1,
                    const float* __restrict__ b2,
                    float* __restrict__ copyA, float* __restrict__ jumpA) {
    __shared__ __align__(16) unsigned short Bbuf[2 * 16 * 32 * 16]; // 32 KB: double-buffered B chunk
    __shared__ __align__(16) unsigned short Atile[128 * 40];        // 10 KB: 128 rows x 32 K (pad 8)
    __shared__ __align__(16) unsigned short Hl[128 * 264];          // 66 KB: h / h2 staging
    __shared__ float scaleS[256], offS[256], b1S[256], b2S[256];
    __shared__ unsigned short colS[576];                            // per-chunk gather offsets

    const int gp = blockIdx.y, g = gp / 9, p = gp - g * 9;
    const int k0 = p / 3, v0 = p - k0 * 3;
    const int patchoff = k0 * 7 + v0;
    const int Kg = g_K(g), npos = g_npos(g), chunks = Kg / 32;
    const int tid = threadIdx.x, lane = tid & 31, wave = tid >> 5;
    const int mrow = (wave & 3) * 32;          // wave's row base within tile
    const int nbase = (wave >> 2) * 8;         // wave's first n-tile (of 16)
    const int ncol = nbase * 16;               // wave's column base
    const int rowbase = blockIdx.x * 128;
    const unsigned bb_lds = lds_addr_of(&Bbuf[0]);

    { // fold BN: z_norm = acc*scale + off (b5 folded in)
        int n = tid;
        float inv = rsqrtf(rvar[gp * 256 + n] + 1e-5f);
        float s = inv * gamma[gp * 256 + n];
        scaleS[n] = s;
        offS[n]   = (b5[gp * 256 + n] - rmean[gp * 256 + n]) * s + beta[gp * 256 + n];
        b1S[n] = b1[gp * 256 + n];
        b2S[n] = b2[gp * 256 + n];
    }
    { // im2col column-offset table (row-invariant): colS[k] = c*49 + (i*7+j)
        int dofftab[9];
        #pragma unroll
        for (int q = 0; q < 9; ++q) {
            int o5 = OFF5[g][q];
            dofftab[q] = (o5 / 5) * 7 + (o5 - (o5 / 5) * 5);
        }
        for (int t = tid; t < Kg; t += 256) {
            int c, pos;
            if (npos == 5) { c = t / 5; pos = t - c * 5; } else { c = t / 9; pos = t - c * 9; }
            colS[t] = (unsigned short)(c * 49 + dofftab[pos]);
        }
    }

    const int aoff = (lane < 16) ? 0 : 16;  // A-fragment lane half-offset
    const int arow16 = lane & 15;

    // ---------------- stage 1: z = patches x W5m^T (packed K) ----------------
    const unsigned short* w5src = w5b + g_base(g) + (size_t)p * Kg * 256;
    v8f acc[2][8] = {};
    if (wave == 0) tdm_issue(w5src, bb_lds, 16384);                  // chunk 0 in flight
    for (int ch = 0; ch < chunks; ++ch) {
        __syncthreads();  // prior readers of the other buffer are done; colS published
        if (wave == 0 && ch + 1 < chunks)
            tdm_issue(w5src + (size_t)(ch + 1) * 8192, bb_lds + ((ch + 1) & 1) * 16384, 16384);
        { // im2col gather of 32 packed-K columns for 128 rows (overlaps TDM)
            int r = tid & 127, seg = tid >> 7;
            const unsigned short* xrow = xb + (size_t)(rowbase + r) * 3136 + patchoff;
            unsigned short* adst = &Atile[r * 40 + seg * 16];
            int kb = ch * 32 + seg * 16;
            #pragma unroll
            for (int e = 0; e < 16; ++e) adst[e] = xrow[colS[kb + e]];
        }
        if (wave == 0) {
            if (ch + 1 < chunks) __builtin_amdgcn_s_wait_tensorcnt(1); // chunk ch landed
            else                 __builtin_amdgcn_s_wait_tensorcnt(0);
        }
        __syncthreads();  // publish Bbuf[ch&1] + Atile
        gemm2x8(acc, &Atile[(mrow + arow16) * 40 + aoff],
                     &Atile[(mrow + 16 + arow16) * 40 + aoff],
                     &Bbuf[(ch & 1) * 8192], lane, nbase);
    }
    // BN + Mish -> Hl (32 rows x 128 cols per wave)
    #pragma unroll
    for (int t = 0; t < 8; ++t) {
        int nl = ncol + t * 16 + arow16;
        float s = scaleS[nl], o = offS[nl];
        #pragma unroll
        for (int mf = 0; mf < 2; ++mf) {
            #pragma unroll
            for (int r = 0; r < 8; ++r) {
                int row = mrow + mf * 16 + ((lane < 16) ? r : r + 8);
                Hl[row * 264 + nl] = (unsigned short)f2bf(mishf(acc[mf][t][r] * s + o));
            }
        }
    }

    // ---------------- stage 2: h x W1^T ----------------
    const unsigned short* w1src = w1b + (size_t)gp * 65536;
    v8f acc2[2][8] = {};
    if (wave == 0) tdm_issue(w1src, bb_lds, 16384);
    for (int ch = 0; ch < 8; ++ch) {
        __syncthreads();  // also publishes Hl on first iteration
        if (wave == 0) {
            if (ch + 1 < 8) {
                tdm_issue(w1src + (size_t)(ch + 1) * 8192, bb_lds + ((ch + 1) & 1) * 16384, 16384);
                __builtin_amdgcn_s_wait_tensorcnt(1);
            } else {
                __builtin_amdgcn_s_wait_tensorcnt(0);
            }
        }
        __syncthreads();
        gemm2x8(acc2, &Hl[(mrow + arow16) * 264 + ch * 32 + aoff],
                      &Hl[(mrow + 16 + arow16) * 264 + ch * 32 + aoff],
                      &Bbuf[(ch & 1) * 8192], lane, nbase);
    }
    __syncthreads();  // all waves done reading h before h2 overwrite (rows shared across column-waves)
    #pragma unroll
    for (int t = 0; t < 8; ++t) {
        int nl = ncol + t * 16 + arow16;
        float bb = b1S[nl];
        #pragma unroll
        for (int mf = 0; mf < 2; ++mf) {
            #pragma unroll
            for (int r = 0; r < 8; ++r) {
                int row = mrow + mf * 16 + ((lane < 16) ? r : r + 8);
                Hl[row * 264 + nl] = (unsigned short)f2bf(mishf(acc2[mf][t][r] + bb));
            }
        }
    }

    // ---------------- stage 3: h2 x W2^T, reduce over (g,p) ----------------
    const unsigned short* w2src = w2b + (size_t)gp * 65536;
    v8f acc3[2][8] = {};
    if (wave == 0) tdm_issue(w2src, bb_lds, 16384);
    for (int ch = 0; ch < 8; ++ch) {
        __syncthreads();  // also publishes h2 on first iteration
        if (wave == 0) {
            if (ch + 1 < 8) {
                tdm_issue(w2src + (size_t)(ch + 1) * 8192, bb_lds + ((ch + 1) & 1) * 16384, 16384);
                __builtin_amdgcn_s_wait_tensorcnt(1);
            } else {
                __builtin_amdgcn_s_wait_tensorcnt(0);
            }
        }
        __syncthreads();
        gemm2x8(acc3, &Hl[(mrow + arow16) * 264 + ch * 32 + aoff],
                      &Hl[(mrow + 16 + arow16) * 264 + ch * 32 + aoff],
                      &Bbuf[(ch & 1) * 8192], lane, nbase);
    }
    float* dst = (g == 0) ? copyA : jumpA;
    #pragma unroll
    for (int t = 0; t < 8; ++t) {
        int nl = ncol + t * 16 + arow16;
        float bb = b2S[nl];
        #pragma unroll
        for (int mf = 0; mf < 2; ++mf) {
            #pragma unroll
            for (int r = 0; r < 8; ++r) {
                int row = rowbase + mrow + mf * 16 + ((lane < 16) ? r : r + 8);
                unsafeAtomicAdd(&dst[(size_t)row * 256 + nl], acc3[mf][t][r] + bb);
            }
        }
    }
}

__global__ void final_mul_kernel(const float* __restrict__ copyA, const float* __restrict__ jumpA,
                                 float* __restrict__ out) {
    int i = blockIdx.x * 256 + threadIdx.x;
    out[i] = copyA[i] * jumpA[i];
}

// ---------------- host launcher ----------------
extern "C" void kernel_launch(void* const* d_in, const int* in_sizes, int n_in,
                              void* d_out, int out_size, void* d_ws, size_t ws_size,
                              hipStream_t stream) {
    const float* x     = (const float*)d_in[0];
    const float* W5    = (const float*)d_in[1];
    const float* b5    = (const float*)d_in[2];
    const float* gamma = (const float*)d_in[3];
    const float* beta  = (const float*)d_in[4];
    const float* rmean = (const float*)d_in[5];
    const float* rvar  = (const float*)d_in[6];
    const float* W1    = (const float*)d_in[7];
    const float* b1    = (const float*)d_in[8];
    const float* W2    = (const float*)d_in[9];
    const float* b2    = (const float*)d_in[10];
    float* out = (float*)d_out;

    char* ws = (char*)d_ws;
    unsigned short* xb  = (unsigned short*)(ws + XB_OFF);
    unsigned short* w5b = (unsigned short*)(ws + W5B_OFF);
    unsigned short* w1b = (unsigned short*)(ws + W1B_OFF);
    unsigned short* w2b = (unsigned short*)(ws + W2B_OFF);
    float* copyA = (float*)(ws + COPY_OFF);
    float* jumpA = (float*)(ws + JUMP_OFF);

    zero_acc_kernel<<<4096, 256, 0, stream>>>(copyA, jumpA);
    prep_x_kernel<<<12544, 256, 0, stream>>>((const float4*)x, (uint2*)xb);
    prep_w5_kernel<<<dim3(576, 27), 256, 0, stream>>>(W5, w5b);
    prep_w12_kernel<<<dim3(256, 27), 256, 0, stream>>>(W1, W2, w1b, w2b);
    fused_branch_kernel<<<dim3(32, 27), 256, 0, stream>>>(
        xb, w5b, w1b, w2b, b5, gamma, beta, rmean, rvar, b1, b2, copyA, jumpA);
    final_mul_kernel<<<4096, 256, 0, stream>>>(copyA, jumpA, out);
}
